// Transformer_72610717106824
// MI455X (gfx1250) — compile-verified, matrix operand order checked by reading, not run on hardware
//
#include <hip/hip_runtime.h>
#include <math.h>

// ---------------------------------------------------------------------------
// Types / helpers
// ---------------------------------------------------------------------------
typedef __attribute__((ext_vector_type(16))) __bf16 v16bf;
typedef __attribute__((ext_vector_type(8)))  float  v8f;
typedef unsigned short u16;

union Frag {
    v16bf v;
    uint4 q[2];
    u16   s[16];
};

__device__ __forceinline__ u16 f2bf(float f) {
    unsigned int u = __float_as_uint(f);
    u += 0x7fffu + ((u >> 16) & 1u);               // round-to-nearest-even
    return (u16)(u >> 16);
}
__device__ __forceinline__ float bf2f(u16 s) {
    return __uint_as_float(((unsigned int)s) << 16);
}
__device__ __forceinline__ v8f wmma_bf16(const Frag& a, const Frag& b, v8f c) {
    return __builtin_amdgcn_wmma_f32_16x16x32_bf16(false, a.v, false, b.v,
                                                   (short)0, c, false, false);
}

#define EP_F32    0
#define EP_BF16   1
#define EP_RELU   2
#define EP_BF16_T 3   // bf16 output stored transposed: C[col*M + row]

// ---------------------------------------------------------------------------
// Tensor Data Mover (TDM) support — guarded; falls back to manual staging.
// Builds a 2-D tile D# per CDNA5 ISA (CDNA5 ISA) and issues
// tensor_load_to_lds from one wave, tracked with TENSORcnt.
// ---------------------------------------------------------------------------
#if defined(__has_builtin)
#  if __has_builtin(__builtin_amdgcn_tensor_load_to_lds) && \
      __has_builtin(__builtin_amdgcn_s_wait_tensorcnt)
#    define HAVE_TDM 1
#  endif
#endif
#ifndef HAVE_TDM
#  define HAVE_TDM 0
#endif

#if HAVE_TDM
typedef __attribute__((ext_vector_type(4))) unsigned int u32x4;
typedef __attribute__((ext_vector_type(4))) int          i32x4;
typedef __attribute__((ext_vector_type(8))) int          i32x8;

// 2-D tile of 2-byte elements: tile_d0 (contiguous) x tile_d1 rows,
// row stride = stride0 elements, from gaddr into LDS byte offset lds_off.
__device__ __forceinline__ void tdm_load_2d(unsigned int lds_off,
                                            const void* gaddr,
                                            unsigned int tensor_d0,
                                            unsigned int tensor_d1,
                                            unsigned int tile_d0,
                                            unsigned int tile_d1,
                                            unsigned int stride0) {
    unsigned long long ga = (unsigned long long)gaddr;
    u32x4 g0;
    g0[0] = 1u;                                   // count=1, user descriptor
    g0[1] = lds_off;                              // lds_addr (bytes)
    g0[2] = (unsigned int)ga;                     // global_addr[31:0]
    g0[3] = (unsigned int)((ga >> 32) & 0x1FFFFFFu) | (2u << 30); // [56:32]|type=2
    i32x8 g1;
    g1[0] = (int)(1u << 16);                      // data_size=1 (2 bytes)
    g1[1] = (int)((tensor_d0 & 0xFFFFu) << 16);   // tensor_dim0[15:0]
    g1[2] = (int)((tensor_d0 >> 16) | ((tensor_d1 & 0xFFFFu) << 16));
    g1[3] = (int)((tensor_d1 >> 16) | (tile_d0 << 16));   // tile_dim0
    g1[4] = (int)tile_d1;                         // tile_dim1 (tile_dim2=0)
    g1[5] = (int)stride0;                         // tensor_dim0_stride[31:0]
    g1[6] = 0;                                    // stride0[47:32]|dim1_stride lo
    g1[7] = 0;
    i32x4 z4 = {0, 0, 0, 0};                      // groups 2/3: dims unused
#if __has_include(<hip/amd_detail/amd_gfx1250_TDM.h>)
    i32x8 z8 = {0, 0, 0, 0, 0, 0, 0, 0};
    __builtin_amdgcn_tensor_load_to_lds(g0, g1, z4, z4, z8, 0);
#else
    __builtin_amdgcn_tensor_load_to_lds(g0, g1, z4, z4, 0);
#endif
}
__device__ __forceinline__ unsigned int lds_off_of(const void* p) {
    return (unsigned int)(unsigned long long)p;   // low 32b = LDS byte offset
}
#endif

// ---------------------------------------------------------------------------
// fp32 [L][K][N] -> bf16 [L][N][K] transpose (weights, once per launch)
// ---------------------------------------------------------------------------
__global__ __launch_bounds__(256) void transpose_to_bf16_kernel(
    const float* __restrict__ src, u16* __restrict__ dst,
    int K, int N, size_t total) {
    size_t i = (size_t)blockIdx.x * blockDim.x + threadIdx.x;
    size_t stride = (size_t)gridDim.x * blockDim.x;
    size_t mat_sz = (size_t)K * N;
    for (; i < total; i += stride) {
        size_t mat = i / mat_sz;
        size_t rem = i - mat * mat_sz;
        int k = (int)(rem / N);
        int n = (int)(rem - (size_t)k * N);
        dst[mat * mat_sz + (size_t)n * K + k] = f2bf(src[i]);
    }
}

// ---------------------------------------------------------------------------
// Embedding gather + scale + sinusoidal positional encoding -> bf16
// ---------------------------------------------------------------------------
__global__ __launch_bounds__(256) void embed_kernel(
    const int* __restrict__ tok, const float* __restrict__ emb,
    u16* __restrict__ out, int S, int d, float scale) {
    int i = blockIdx.x * blockDim.x + threadIdx.x;   // over B*S*d
    int row = i / d, f = i - row * d;
    int pos = row % S;
    int t = tok[row];
    float dt  = __expf((float)(f & ~1) * (-9.2103403719761836f / (float)d));
    float ang = (float)pos * dt;
    float pe  = (f & 1) ? __cosf(ang) : __sinf(ang);
    out[(size_t)row * d + f] = f2bf(emb[(size_t)t * d + f] * scale + pe);
}

// ---------------------------------------------------------------------------
// Tiled bf16 GEMM: C[M,N] = A[M,K] @ Bt^T + bias, WMMA f32 accumulate.
// A is [M,K] row-major; Bt is the PRE-TRANSPOSED weight [N,K] row-major,
// so both LDS tiles are contiguous-along-K and stage via TDM (2-D tiles).
// Block 256 threads = 8 waves (4 along M x 2 along N). Tile 128x128x32.
// ---------------------------------------------------------------------------
#define BM 128
#define BN 128
#define BK 32

__global__ __launch_bounds__(256) void gemm_bf16_kernel(
    const u16* __restrict__ A, const u16* __restrict__ Bt,
    const float* __restrict__ bias, void* __restrict__ C,
    int M, int N, int K, int epilogue) {
    __shared__ __align__(16) u16 As[BM * BK];       // [m][k]
    __shared__ __align__(16) u16 Bs[BN * BK];       // [n][k]

    int tid  = threadIdx.x;
    int lane = tid & 31, wave = tid >> 5;
    int wm   = wave & 3,  wn  = wave >> 2;          // 4x2 wave grid
    int m0   = blockIdx.y * BM, n0 = blockIdx.x * BN;
    int mrow = lane & 15, half = lane >> 4;

    v8f acc[2][4];
    #pragma unroll
    for (int i = 0; i < 2; ++i)
        #pragma unroll
        for (int j = 0; j < 4; ++j)
            #pragma unroll
            for (int r = 0; r < 8; ++r) acc[i][j][r] = 0.f;

    for (int k0 = 0; k0 < K; k0 += BK) {
#if HAVE_TDM
        if (wave == 0) {
            tdm_load_2d(lds_off_of(&As[0]), A  + (size_t)m0 * K + k0,
                        (unsigned)K, (unsigned)M, BK, BM, (unsigned)K);
            tdm_load_2d(lds_off_of(&Bs[0]), Bt + (size_t)n0 * K + k0,
                        (unsigned)K, (unsigned)N, BK, BN, (unsigned)K);
            __builtin_amdgcn_s_wait_tensorcnt(0);
        }
#else
        #pragma unroll
        for (int it = 0; it < 2; ++it) {
            int idx = it * 256 + tid;               // 512 chunks of 8 halves
            int r = idx >> 2, cc = (idx & 3) * 8;
            *(uint4*)&As[r * BK + cc] =
                *(const uint4*)&A[(size_t)(m0 + r) * K + k0 + cc];
        }
        #pragma unroll
        for (int it = 0; it < 2; ++it) {
            int idx = it * 256 + tid;
            int n = idx >> 2, cc = (idx & 3) * 8;
            *(uint4*)&Bs[n * BK + cc] =
                *(const uint4*)&Bt[(size_t)(n0 + n) * K + k0 + cc];
        }
#endif
        if (k0 + BK < K) {   // hint next tiles toward GL2 (global_prefetch_b8)
            __builtin_prefetch(&A[(size_t)(m0 + (tid >> 1)) * K + k0 + BK], 0, 1);
            __builtin_prefetch(&Bt[(size_t)(n0 + (tid >> 1)) * K + k0 + BK], 0, 1);
        }
        __syncthreads();

        Frag aF[2], bF[4];
        #pragma unroll
        for (int t = 0; t < 2; ++t) {
            int r = wm * 32 + t * 16 + mrow;
            aF[t].q[0] = *(const uint4*)&As[r * BK + 8 * half];
            aF[t].q[1] = *(const uint4*)&As[r * BK + 16 + 8 * half];
        }
        #pragma unroll
        for (int t = 0; t < 4; ++t) {
            int n = wn * 64 + t * 16 + mrow;
            bF[t].q[0] = *(const uint4*)&Bs[n * BK + 8 * half];
            bF[t].q[1] = *(const uint4*)&Bs[n * BK + 16 + 8 * half];
        }
        #pragma unroll
        for (int i = 0; i < 2; ++i)
            #pragma unroll
            for (int j = 0; j < 4; ++j)
                acc[i][j] = wmma_bf16(aF[i], bF[j], acc[i][j]);
        __syncthreads();
    }

    // epilogue: C layout per ISA: lane = N-col, VGPR r -> M = r + 8*half
    #pragma unroll
    for (int i = 0; i < 2; ++i) {
        #pragma unroll
        for (int j = 0; j < 4; ++j) {
            int col = n0 + wn * 64 + j * 16 + mrow;
            float bv = bias ? bias[col] : 0.f;
            #pragma unroll
            for (int r = 0; r < 8; ++r) {
                int row = m0 + wm * 32 + i * 16 + half * 8 + r;
                float v = acc[i][j][r] + bv;
                if (epilogue == EP_RELU && v < 0.f) v = 0.f;
                if (epilogue == EP_F32)
                    ((float*)C)[(size_t)row * N + col] = v;
                else if (epilogue == EP_BF16_T)
                    ((u16*)C)[(size_t)col * M + row] = f2bf(v);
                else
                    ((u16*)C)[(size_t)row * N + col] = f2bf(v);
            }
        }
    }
}

// ---------------------------------------------------------------------------
// Flash-style attention. One head slice (dk=64) per blockIdx.y, batch per z.
// Block = 128 threads = 4 waves; each wave owns 16 queries. 32-key chunks.
// K given row-major [tokens][d]; V given PRE-TRANSPOSED [d][tokens] so both
// chunks stage via TDM / contiguous b128 copies (no LDS-side transpose).
// ---------------------------------------------------------------------------
__global__ __launch_bounds__(128) void attention_kernel(
    const u16* __restrict__ Q, const u16* __restrict__ Kin,
    const u16* __restrict__ VtG, u16* __restrict__ O,
    const int* __restrict__ keyTok, int Sq, int Sk, int d, int vs,
    int causal, float scale) {
    const int DK = 64;
    __shared__ __align__(16) u16 Ks[32 * DK];       // [key][dk]
    __shared__ __align__(16) u16 Vt[DK * 32];       // [dk][key]
    __shared__ __align__(16) u16 Ps[4][16 * 32];    // per-wave P scratch

    int tid  = threadIdx.x;
    int lane = tid & 31, wave = tid >> 5;
    int b = blockIdx.z, h = blockIdx.y;
    int q0 = blockIdx.x * 64 + wave * 16;
    int mrow = lane & 15, hf = lane >> 4;

    // Q fragments (invariant over key loop): 2 frags covering dk 0..63
    Frag qF[2];
    {
        const u16* qp = Q + ((size_t)(b * Sq + q0 + mrow)) * d + h * DK;
        #pragma unroll
        for (int c = 0; c < 2; ++c) {
            qF[c].q[0] = *(const uint4*)(qp + c * 32 + 8 * hf);
            qF[c].q[1] = *(const uint4*)(qp + c * 32 + 16 + 8 * hf);
        }
    }

    float mcur[8], lsum[8];
    v8f out[4];
    #pragma unroll
    for (int r = 0; r < 8; ++r) { mcur[r] = -1e30f; lsum[r] = 0.f; }
    #pragma unroll
    for (int j = 0; j < 4; ++j)
        #pragma unroll
        for (int r = 0; r < 8; ++r) out[j][r] = 0.f;

    for (int k0 = 0; k0 < Sk; k0 += 32) {
        // ---- stage K chunk [32][64] and Vt chunk [64][32] ----
#if HAVE_TDM
        if (wave == 0) {
            tdm_load_2d(lds_off_of(&Ks[0]),
                        Kin + ((size_t)(b * Sk + k0)) * d + h * DK,
                        (unsigned)d, (unsigned)Sk, DK, 32, (unsigned)d);
            tdm_load_2d(lds_off_of(&Vt[0]),
                        VtG + (size_t)(h * DK) * vs + (size_t)b * Sk + k0,
                        (unsigned)vs, (unsigned)d, 32, DK, (unsigned)vs);
            __builtin_amdgcn_s_wait_tensorcnt(0);
        }
#else
        #pragma unroll
        for (int it = 0; it < 2; ++it) {
            int idx = it * 128 + tid;               // 256 chunks of 8 halves
            int kr = idx >> 3, cc = (idx & 7) * 8;
            *(uint4*)&Ks[kr * DK + cc] =
                *(const uint4*)&Kin[((size_t)(b * Sk + k0 + kr)) * d + h * DK + cc];
        }
        #pragma unroll
        for (int it = 0; it < 2; ++it) {
            int idx = it * 128 + tid;               // 256 chunks of 8 halves
            int r = idx >> 2, cc = (idx & 3) * 8;
            *(uint4*)&Vt[r * 32 + cc] =
                *(const uint4*)&VtG[(size_t)(h * DK + r) * vs + (size_t)b * Sk + k0 + cc];
        }
#endif
        __syncthreads();

        // logits: two 16x16 tiles, each = 2 chained WMMAs over dk=64
        float logit[2][8];
        float rmax[8];
        #pragma unroll
        for (int r = 0; r < 8; ++r) rmax[r] = -1e30f;
        #pragma unroll
        for (int t = 0; t < 2; ++t) {
            Frag kF[2];
            int n = t * 16 + mrow;                  // key column (local)
            #pragma unroll
            for (int c = 0; c < 2; ++c) {
                kF[c].q[0] = *(const uint4*)&Ks[n * DK + c * 32 + 8 * hf];
                kF[c].q[1] = *(const uint4*)&Ks[n * DK + c * 32 + 16 + 8 * hf];
            }
            v8f lg;
            #pragma unroll
            for (int r = 0; r < 8; ++r) lg[r] = 0.f;
            lg = wmma_bf16(qF[0], kF[0], lg);
            lg = wmma_bf16(qF[1], kF[1], lg);
            int key = k0 + n;
            bool padm = (keyTok[b * Sk + key] == 1);
            #pragma unroll
            for (int r = 0; r < 8; ++r) {
                int qg = q0 + hf * 8 + r;
                float v = lg[r] * scale;
                if (padm || (causal && key > qg)) v = -1e30f;
                logit[t][r] = v;
                rmax[r] = fmaxf(rmax[r], v);
            }
        }
        // row max across the 16 lanes of each half-wave
        #pragma unroll
        for (int r = 0; r < 8; ++r) {
            float v = rmax[r];
            for (int off = 1; off < 16; off <<= 1)
                v = fmaxf(v, __shfl_xor(v, off, 32));
            rmax[r] = v;
        }
        // online softmax update
        float corr[8], mnew[8], psum[8];
        #pragma unroll
        for (int r = 0; r < 8; ++r) {
            mnew[r] = fmaxf(mcur[r], rmax[r]);
            corr[r] = __expf(mcur[r] - mnew[r]);
            mcur[r] = mnew[r];
            psum[r] = 0.f;
        }
        #pragma unroll
        for (int t = 0; t < 2; ++t)
            #pragma unroll
            for (int r = 0; r < 8; ++r) {
                float p = (logit[t][r] <= -1e29f) ? 0.f
                                                  : __expf(logit[t][r] - mnew[r]);
                psum[r] += p;
                Ps[wave][(hf * 8 + r) * 32 + t * 16 + mrow] = f2bf(p);
            }
        #pragma unroll
        for (int r = 0; r < 8; ++r) {
            float s = psum[r];
            for (int off = 1; off < 16; off <<= 1) s += __shfl_xor(s, off, 32);
            lsum[r] = lsum[r] * corr[r] + s;
        }
        #pragma unroll
        for (int j = 0; j < 4; ++j)
            #pragma unroll
            for (int r = 0; r < 8; ++r) out[j][r] *= corr[r];

        // P @ V : 4 WMMAs (out dims in 16-col tiles)
        Frag pF;
        pF.q[0] = *(const uint4*)&Ps[wave][mrow * 32 + 8 * hf];
        pF.q[1] = *(const uint4*)&Ps[wave][mrow * 32 + 16 + 8 * hf];
        #pragma unroll
        for (int j = 0; j < 4; ++j) {
            Frag vF;
            int n = j * 16 + mrow;                  // out-dim column
            vF.q[0] = *(const uint4*)&Vt[n * 32 + 8 * hf];
            vF.q[1] = *(const uint4*)&Vt[n * 32 + 16 + 8 * hf];
            out[j] = wmma_bf16(pF, vF, out[j]);
        }
        __syncthreads();
    }

    // finalize: divide by l (fully masked rows -> 0, matches nan_to_num)
    #pragma unroll
    for (int j = 0; j < 4; ++j)
        #pragma unroll
        for (int r = 0; r < 8; ++r) {
            float l = lsum[r];
            float v = (l > 0.f) ? out[j][r] / l : 0.f;
            int qg = q0 + hf * 8 + r;
            O[((size_t)(b * Sq + qg)) * d + h * DK + j * 16 + mrow] = f2bf(v);
        }
}

// ---------------------------------------------------------------------------
// Residual add + LayerNorm. One block per row (d=512), 256 threads.
// gin (f32, may be null) + resid (bf16, may be null) -> LN -> bf16
// ---------------------------------------------------------------------------
__global__ __launch_bounds__(256) void add_ln_kernel(
    const float* __restrict__ gin, const u16* __restrict__ resid,
    const float* __restrict__ g, const float* __restrict__ b,
    u16* __restrict__ out, int d) {
    __shared__ float ws[8], wq[8];
    int row = blockIdx.x;
    int it = d >> 8;                                 // = 2 for d=512
    float v[8];
    float s = 0.f, sq = 0.f;
    for (int i = 0; i < it; ++i) {
        int c = i * 256 + threadIdx.x;
        float x = 0.f;
        if (gin)   x += gin[(size_t)row * d + c];
        if (resid) x += bf2f(resid[(size_t)row * d + c]);
        v[i] = x; s += x; sq += x * x;
    }
    for (int off = 1; off < 32; off <<= 1) {
        s  += __shfl_xor(s, off, 32);
        sq += __shfl_xor(sq, off, 32);
    }
    int wave = threadIdx.x >> 5, lane = threadIdx.x & 31;
    if (lane == 0) { ws[wave] = s; wq[wave] = sq; }
    __syncthreads();
    s = 0.f; sq = 0.f;
    for (int w = 0; w < 8; ++w) { s += ws[w]; sq += wq[w]; }
    float mu  = s / (float)d;
    float var = sq / (float)d - mu * mu;
    float inv = rsqrtf(var + 1e-5f);
    for (int i = 0; i < it; ++i) {
        int c = i * 256 + threadIdx.x;
        out[(size_t)row * d + c] = f2bf((v[i] - mu) * inv * g[c] + b[c]);
    }
}

// ---------------------------------------------------------------------------
// Host orchestration
// ---------------------------------------------------------------------------
extern "C" void kernel_launch(void* const* d_in, const int* in_sizes, int n_in,
                              void* d_out, int out_size, void* d_ws, size_t ws_size,
                              hipStream_t stream) {
    const int B = 8, S = 512, T = 512, d = 512, dff = 2048, N = 6, V = 32000, H = 8;
    const int M = B * S;
    (void)in_sizes; (void)n_in; (void)out_size;

    const int*   src        = (const int*)d_in[0];
    const int*   tgt        = (const int*)d_in[1];
    const float* src_emb    = (const float*)d_in[2];
    const float* tgt_emb    = (const float*)d_in[3];
    const float* enc_attn_w = (const float*)d_in[4];
    const float* enc_attn_b = (const float*)d_in[5];
    const float* enc_ffn_w1 = (const float*)d_in[6];
    const float* enc_ffn_b1 = (const float*)d_in[7];
    const float* enc_ffn_w2 = (const float*)d_in[8];
    const float* enc_ffn_b2 = (const float*)d_in[9];
    const float* enc_ln_g   = (const float*)d_in[10];
    const float* enc_ln_b   = (const float*)d_in[11];
    const float* enc_norm_g = (const float*)d_in[12];
    const float* enc_norm_b = (const float*)d_in[13];
    const float* dec_attn_w = (const float*)d_in[14];
    const float* dec_attn_b = (const float*)d_in[15];
    const float* dec_ffn_w1 = (const float*)d_in[16];
    const float* dec_ffn_b1 = (const float*)d_in[17];
    const float* dec_ffn_w2 = (const float*)d_in[18];
    const float* dec_ffn_b2 = (const float*)d_in[19];
    const float* dec_ln_g   = (const float*)d_in[20];
    const float* dec_ln_b   = (const float*)d_in[21];
    const float* dec_norm_g = (const float*)d_in[22];
    const float* dec_norm_b = (const float*)d_in[23];
    const float* gen_w      = (const float*)d_in[24];
    const float* gen_b      = (const float*)d_in[25];

    char* base = (char*)d_ws;
    size_t off = 0;
    auto alloc = [&](size_t bytes) -> void* {
        void* p = base + off;
        off += (bytes + 255) & ~(size_t)255;
        return p;
    };
    // all weight copies below are PRE-TRANSPOSED: [L][N][K] bf16
    u16* wEncAttn = (u16*)alloc((size_t)N * 4 * d * d * 2);
    u16* wEncF1   = (u16*)alloc((size_t)N * d * dff * 2);
    u16* wEncF2   = (u16*)alloc((size_t)N * dff * d * 2);
    u16* wDecAttn = (u16*)alloc((size_t)N * 8 * d * d * 2);
    u16* wDecF1   = (u16*)alloc((size_t)N * d * dff * 2);
    u16* wDecF2   = (u16*)alloc((size_t)N * dff * d * 2);
    u16* wGen     = (u16*)alloc((size_t)d * V * 2);
    u16* xb    = (u16*)alloc((size_t)M * d * 2);
    u16* qb    = (u16*)alloc((size_t)M * d * 2);
    u16* kb    = (u16*)alloc((size_t)M * d * 2);
    u16* vtb   = (u16*)alloc((size_t)M * d * 2);   // V projection, transposed [d][M]
    u16* attnb = (u16*)alloc((size_t)M * d * 2);
    u16* memb  = (u16*)alloc((size_t)M * d * 2);
    u16* yb    = (u16*)alloc((size_t)M * d * 2);
    u16* y2b   = (u16*)alloc((size_t)M * d * 2);
    u16* hb    = (u16*)alloc((size_t)M * dff * 2);
    float* gf  = (float*)alloc((size_t)M * d * 4);
    if (off > ws_size) return;   // workspace too small; do nothing deterministic

    auto convT = [&](const float* s, u16* t, int K, int Nn, int L) {
        size_t total = (size_t)L * K * Nn;
        int blocks = (int)((total + 2047) / 2048);
        if (blocks > 16384) blocks = 16384;
        transpose_to_bf16_kernel<<<blocks, 256, 0, stream>>>(s, t, K, Nn, total);
    };
    auto gemm = [&](const u16* A, const u16* Bw, const float* bias, void* C,
                    int Mi, int Ni, int Ki, int ep) {
        dim3 grid(Ni / BN, Mi / BM);
        gemm_bf16_kernel<<<grid, 256, 0, stream>>>(A, Bw, bias, C, Mi, Ni, Ki, ep);
    };
    auto attn = [&](const u16* Qp, const u16* Kp, const u16* Vtp, u16* Op,
                    const int* keyTok, int Sq, int Sk, int causal) {
        dim3 grid(Sq / 64, H, B);
        attention_kernel<<<grid, 128, 0, stream>>>(Qp, Kp, Vtp, Op, keyTok,
                                                   Sq, Sk, d, M, causal, 0.125f);
    };
    auto lnorm = [&](const float* gi, const u16* res, const float* g,
                     const float* b, u16* o) {
        add_ln_kernel<<<M, 256, 0, stream>>>(gi, res, g, b, o, d);
    };

    // ---- convert + transpose weights to bf16 [N][K] ----
    convT(enc_attn_w, wEncAttn, d, d, N * 4);
    convT(enc_ffn_w1, wEncF1,   d, dff, N);
    convT(enc_ffn_w2, wEncF2,   dff, d, N);
    convT(dec_attn_w, wDecAttn, d, d, N * 8);
    convT(dec_ffn_w1, wDecF1,   d, dff, N);
    convT(dec_ffn_w2, wDecF2,   dff, d, N);
    convT(gen_w,      wGen,     d, V, 1);

    const float scale_emb = sqrtf((float)d);

    // ---- encoder ----
    embed_kernel<<<(M * d) / 256, 256, 0, stream>>>(src, src_emb, xb, S, d, scale_emb);
    for (int i = 0; i < N; ++i) {
        const u16* wq = wEncAttn + (size_t)(i * 4 + 0) * d * d;
        const u16* wk = wEncAttn + (size_t)(i * 4 + 1) * d * d;
        const u16* wv = wEncAttn + (size_t)(i * 4 + 2) * d * d;
        const u16* wo = wEncAttn + (size_t)(i * 4 + 3) * d * d;
        gemm(xb, wq, enc_attn_b + (size_t)(i * 4 + 0) * d, qb, M, d, d, EP_BF16);
        gemm(xb, wk, enc_attn_b + (size_t)(i * 4 + 1) * d, kb, M, d, d, EP_BF16);
        gemm(xb, wv, enc_attn_b + (size_t)(i * 4 + 2) * d, vtb, M, d, d, EP_BF16_T);
        attn(qb, kb, vtb, attnb, src, S, S, 0);
        gemm(attnb, wo, enc_attn_b + (size_t)(i * 4 + 3) * d, gf, M, d, d, EP_F32);
        lnorm(gf, xb, enc_ln_g + (size_t)(i * 2 + 0) * d, enc_ln_b + (size_t)(i * 2 + 0) * d, xb);
        gemm(xb, wEncF1 + (size_t)i * d * dff, enc_ffn_b1 + (size_t)i * dff, hb, M, dff, d, EP_RELU);
        gemm(hb, wEncF2 + (size_t)i * dff * d, enc_ffn_b2 + (size_t)i * d, gf, M, d, dff, EP_F32);
        lnorm(gf, xb, enc_ln_g + (size_t)(i * 2 + 1) * d, enc_ln_b + (size_t)(i * 2 + 1) * d, xb);
    }
    lnorm(nullptr, xb, enc_norm_g, enc_norm_b, memb);

    // ---- decoder ----
    embed_kernel<<<(M * d) / 256, 256, 0, stream>>>(tgt, tgt_emb, yb, T, d, scale_emb);
    for (int i = 0; i < N; ++i) {
        // self-attention (causal + tgt padding)
        gemm(yb, wDecAttn + (size_t)(i * 8 + 0) * d * d, dec_attn_b + (size_t)(i * 8 + 0) * d, qb, M, d, d, EP_BF16);
        gemm(yb, wDecAttn + (size_t)(i * 8 + 1) * d * d, dec_attn_b + (size_t)(i * 8 + 1) * d, kb, M, d, d, EP_BF16);
        gemm(yb, wDecAttn + (size_t)(i * 8 + 2) * d * d, dec_attn_b + (size_t)(i * 8 + 2) * d, vtb, M, d, d, EP_BF16_T);
        attn(qb, kb, vtb, attnb, tgt, T, T, 1);
        gemm(attnb, wDecAttn + (size_t)(i * 8 + 3) * d * d, dec_attn_b + (size_t)(i * 8 + 3) * d, gf, M, d, d, EP_F32);
        lnorm(gf, yb, dec_ln_g + (size_t)(i * 3 + 0) * d, dec_ln_b + (size_t)(i * 3 + 0) * d, yb);
        // cross-attention (src padding)
        gemm(yb,   wDecAttn + (size_t)(i * 8 + 4) * d * d, dec_attn_b + (size_t)(i * 8 + 4) * d, qb, M, d, d, EP_BF16);
        gemm(memb, wDecAttn + (size_t)(i * 8 + 5) * d * d, dec_attn_b + (size_t)(i * 8 + 5) * d, kb, M, d, d, EP_BF16);
        gemm(memb, wDecAttn + (size_t)(i * 8 + 6) * d * d, dec_attn_b + (size_t)(i * 8 + 6) * d, vtb, M, d, d, EP_BF16_T);
        attn(qb, kb, vtb, attnb, src, T, S, 0);
        gemm(attnb, wDecAttn + (size_t)(i * 8 + 7) * d * d, dec_attn_b + (size_t)(i * 8 + 7) * d, gf, M, d, d, EP_F32);
        lnorm(gf, yb, dec_ln_g + (size_t)(i * 3 + 1) * d, dec_ln_b + (size_t)(i * 3 + 1) * d, yb);
        // FFN
        gemm(yb, wDecF1 + (size_t)i * d * dff, dec_ffn_b1 + (size_t)i * dff, hb, M, dff, d, EP_RELU);
        gemm(hb, wDecF2 + (size_t)i * dff * d, dec_ffn_b2 + (size_t)i * d, gf, M, d, dff, EP_F32);
        lnorm(gf, yb, dec_ln_g + (size_t)(i * 3 + 2) * d, dec_ln_b + (size_t)(i * 3 + 2) * d, yb);
    }
    lnorm(nullptr, yb, dec_norm_g, dec_norm_b, y2b);

    // ---- generator: [M,512] @ W^T[32000,512] + bias -> d_out (f32) ----
    gemm(y2b, wGen, gen_b, d_out, M, V, d, EP_F32);
}
